// CentroidLayer_70652212019778
// MI455X (gfx1250) — compile-verified, matrix operand order checked by reading, not run on hardware
//
#include <hip/hip_runtime.h>
#include <stdint.h>

// Sizes from the reference
#define BQ 131072
#define PN 64
#define DN 256

typedef __attribute__((ext_vector_type(16))) __bf16    v16bf;
typedef __attribute__((ext_vector_type(8)))  float     v8f;
typedef __attribute__((ext_vector_type(4)))  uint32_t  u32x4;

union Frag {
    v16bf          v;
    __bf16         h[16];
    u32x4          q[2];
    uint32_t       d[8];
    unsigned short u[16];
};

// ---------------------------------------------------------------------------
// Prep: normalize centroids -> bf16 [64][256]; raw centroids transposed ->
// bf16 [256][64]; active mask -> float flags. One block per centroid row.
// ---------------------------------------------------------------------------
__global__ __launch_bounds__(256)
void centroid_prep(const float* __restrict__ cent,
                   const unsigned char* __restrict__ act,
                   __bf16* __restrict__ c_norm,
                   __bf16* __restrict__ c_rawT,
                   float* __restrict__ maskf)
{
    __shared__ float red[256];
    const int p = blockIdx.x;
    const int t = threadIdx.x;
    const float x = cent[p * DN + t];
    red[t] = x * x;
    __syncthreads();
    for (int s = 128; s > 0; s >>= 1) {
        if (t < s) red[t] += red[t + s];
        __syncthreads();
    }
    const float inv = 1.0f / fmaxf(sqrtf(red[0]), 1e-12f);
    c_norm[p * DN + t] = (__bf16)(x * inv);   // normalized, row-major [P][D]
    c_rawT[t * PN + p] = (__bf16)x;           // raw, transposed [D][P]
    if (t == 0) maskf[p] = act[p] ? 1.0f : 0.0f;
}

// ---------------------------------------------------------------------------
// Main fused kernel: 8 waves/block, 16 query rows/wave (one WMMA M-tile).
// Single pass over q: build raw-bf16 A fragments while accumulating sum of
// squares; per-row 1/||q|| is applied to the f32 accumulator post-WMMA.
// ---------------------------------------------------------------------------
__global__ __launch_bounds__(256)
void centroid_main(const float* __restrict__ qg,
                   const __bf16* __restrict__ c_norm,
                   const __bf16* __restrict__ c_rawT,
                   const float* __restrict__ maskf,
                   float* __restrict__ ctx_out,
                   float* __restrict__ w_out,
                   float* __restrict__ hard_out)
{
    __shared__ __align__(16) __bf16 w_stage[8][16][PN]; // 16 KB bf16 weights
    __shared__ float mask_s[PN];

    const int tid   = threadIdx.x;
    const int wave  = tid >> 5;
    const int lane  = tid & 31;
    const int half  = lane >> 4;     // which 16-lane half
    const int l16   = lane & 15;
    const int kbase = half * 8;      // A/B 16-bit fragment K-group offset
    const int row0  = blockIdx.x * 128 + wave * 16;

    if (tid < PN) mask_s[tid] = maskf[tid];
    __syncthreads();

    // ---- single pass: build 8 A fragments (raw q -> bf16) + sum of squares.
    // Lane pair (l16, l16+16) covers each element of row l16 exactly once.
    const float* qr = qg + (size_t)(row0 + l16) * DN;
    Frag af[8];
    float ss = 0.f;
    #pragma unroll
    for (int kc = 0; kc < 8; ++kc) {
        const int k0 = kc * 32 + kbase;
        #pragma unroll
        for (int j = 0; j < 8; ++j) {
            const float a = qr[k0 + j];        // K = k0..k0+7
            const float b = qr[k0 + 16 + j];   // K = k0+16..k0+23
            ss = fmaf(a, a, ss);
            ss = fmaf(b, b, ss);
            af[kc].h[j]     = (__bf16)a;
            af[kc].h[8 + j] = (__bf16)b;
        }
    }
    ss += __shfl_xor(ss, 16, 32);              // combine lane-pair halves
    const float inv = 1.0f / fmaxf(sqrtf(ss), 1e-12f);

    // ---- sim_raw = q_raw @ c_norm^T : 4 N-tiles of 16, K = 256 ----
    v8f acc[4];
    #pragma unroll
    for (int pt = 0; pt < 4; ++pt) {
        v8f a = {0.f,0.f,0.f,0.f,0.f,0.f,0.f,0.f};
        const __bf16* crow = c_norm + (size_t)(pt * 16 + l16) * DN; // row of B^T
        #pragma unroll
        for (int kc = 0; kc < 8; ++kc) {
            Frag bf;
            bf.q[0] = *(const u32x4*)(crow + kc * 32 + kbase);
            bf.q[1] = *(const u32x4*)(crow + kc * 32 + kbase + 16);
            a = __builtin_amdgcn_wmma_f32_16x16x32_bf16(false, af[kc].v, false, bf.v,
                                                        (short)0, a, false, false);
        }
        acc[pt] = a;
    }

    // ---- apply per-row 1/||q|| (row of acc element v is v + 8*half) ----
    float invr[8];
    #pragma unroll
    for (int v = 0; v < 8; ++v) invr[v] = __shfl(inv, v + 8 * half, 32);
    #pragma unroll
    for (int pt = 0; pt < 4; ++pt)
        #pragma unroll
        for (int v = 0; v < 8; ++v)
            acc[pt][v] *= invr[v];

    // ---- mask inactive centroids (where-select to -1e9) ----
    #pragma unroll
    for (int pt = 0; pt < 4; ++pt) {
        const bool on = mask_s[pt * 16 + l16] > 0.5f;
        #pragma unroll
        for (int v = 0; v < 8; ++v)
            acc[pt][v] = on ? acc[pt][v] : -1e9f;
    }

    // ---- per-row softmax + argmax over 64 cols ----
    #pragma unroll
    for (int v = 0; v < 8; ++v) {
        float m = acc[0][v];
        int  mi = l16;
        #pragma unroll
        for (int pt = 1; pt < 4; ++pt) {
            float s = acc[pt][v];
            if (s > m) { m = s; mi = pt * 16 + l16; }
        }
        #pragma unroll
        for (int off = 1; off < 16; off <<= 1) {       // butterfly within 16 lanes
            float om = __shfl_xor(m, off, 32);
            int  omi = __shfl_xor(mi, off, 32);
            if (om > m || (om == m && omi < mi)) { m = om; mi = omi; }
        }
        float e[4], sum = 0.f;
        #pragma unroll
        for (int pt = 0; pt < 4; ++pt) { e[pt] = __expf(acc[pt][v] - m); sum += e[pt]; }
        #pragma unroll
        for (int off = 1; off < 16; off <<= 1) sum += __shfl_xor(sum, off, 32);
        const float rs = 1.0f / sum;
        const int r = row0 + v + 8 * half;
        #pragma unroll
        for (int pt = 0; pt < 4; ++pt) {
            const float w = e[pt] * rs;
            w_out[(size_t)r * PN + pt * 16 + l16] = w;
            w_stage[wave][v + 8 * half][pt * 16 + l16] = (__bf16)w;
        }
        if (l16 == 0) hard_out[r] = (float)mi;
    }
    __syncthreads();

    // ---- context = weights @ centroid_raw : A = weights [16x64] from LDS ----
    Frag wa[2];
    const __bf16* wr = &w_stage[wave][l16][0];
    #pragma unroll
    for (int kf = 0; kf < 2; ++kf) {
        wa[kf].q[0] = *(const u32x4*)(wr + kf * 32 + kbase);
        wa[kf].q[1] = *(const u32x4*)(wr + kf * 32 + kbase + 16);
    }
    #pragma unroll
    for (int nt = 0; nt < 16; ++nt) {
        v8f a = {0.f,0.f,0.f,0.f,0.f,0.f,0.f,0.f};
        const __bf16* trow = c_rawT + (size_t)(nt * 16 + l16) * PN; // row of B^T
        #pragma unroll
        for (int kf = 0; kf < 2; ++kf) {
            Frag bf;
            bf.q[0] = *(const u32x4*)(trow + kf * 32 + kbase);
            bf.q[1] = *(const u32x4*)(trow + kf * 32 + kbase + 16);
            a = __builtin_amdgcn_wmma_f32_16x16x32_bf16(false, wa[kf].v, false, bf.v,
                                                        (short)0, a, false, false);
        }
        #pragma unroll
        for (int v = 0; v < 8; ++v)
            ctx_out[(size_t)(row0 + v + 8 * half) * DN + nt * 16 + l16] = a[v];
    }
}

// ---------------------------------------------------------------------------
extern "C" void kernel_launch(void* const* d_in, const int* in_sizes, int n_in,
                              void* d_out, int out_size, void* d_ws, size_t ws_size,
                              hipStream_t stream) {
    (void)in_sizes; (void)n_in; (void)out_size; (void)ws_size;

    const float*         q    = (const float*)d_in[0];         // [B,D] f32
    const float*         cent = (const float*)d_in[1];         // [P,D] f32
    const unsigned char* act  = (const unsigned char*)d_in[2]; // [P] bool

    // workspace layout
    __bf16* c_norm = (__bf16*)d_ws;                          // 32768 B
    __bf16* c_rawT = (__bf16*)((char*)d_ws + 32768);         // 32768 B
    float*  maskf  = (float*)((char*)d_ws + 65536);          //   256 B

    // output layout: context [B*D], weights [B*P], hard [B] (as f32 values)
    float* out   = (float*)d_out;
    float* ctx   = out;
    float* wts   = out + (size_t)BQ * DN;
    float* hardo = wts + (size_t)BQ * PN;

    centroid_prep<<<PN, 256, 0, stream>>>(cent, act, c_norm, c_rawT, maskf);
    centroid_main<<<BQ / 128, 256, 0, stream>>>(q, c_norm, c_rawT, maskf,
                                                ctx, wts, hardo);
}
